// Pointnet_Tracking_19069654794728
// MI455X (gfx1250) — matrix operand order repaired and checked by reading, not compiled, not run on hardware
//
#include <hip/hip_runtime.h>
#include <hip/hip_bf16.h>
#include <math.h>

typedef __attribute__((ext_vector_type(16))) _Float16 v16h;
typedef __attribute__((ext_vector_type(8)))  float    v8f;

#define DEV_B 32

// ---------------------------------------------------------------------------
// Farthest point sampling: one block per batch, matches jax scan semantics:
// out[0]=0; for it>=1: dist=min(dist,d(pts,last)); out[it]=argmax(dist) (first max).
// ---------------------------------------------------------------------------
__global__ void fps_kernel(const float* __restrict__ xyz, int N, int npoint,
                           int* __restrict__ out) {
  int b = blockIdx.x;
  const float* p = xyz + (size_t)b * N * 3;
  int* o = out + (size_t)b * npoint;
  __shared__ float dist[1024];
  __shared__ float rv[256];
  __shared__ int   ri[256];
  __shared__ int   s_last;
  int t = threadIdx.x;
  for (int i = t; i < N; i += 256) dist[i] = 1e10f;
  if (t == 0) { s_last = 0; o[0] = 0; }
  __syncthreads();
  for (int it = 1; it < npoint; ++it) {
    int last = s_last;
    float lx = p[last * 3 + 0], ly = p[last * 3 + 1], lz = p[last * 3 + 2];
    float bv = -1.f; int bi = 0;
    for (int i = t; i < N; i += 256) {
      float dx = p[i * 3 + 0] - lx, dy = p[i * 3 + 1] - ly, dz = p[i * 3 + 2] - lz;
      float d = dx * dx + dy * dy + dz * dz;
      float nd = fminf(dist[i], d);
      dist[i] = nd;
      if (nd > bv || (nd == bv && i < bi)) { bv = nd; bi = i; }
    }
    rv[t] = bv; ri[t] = bi;
    __syncthreads();
    for (int s = 128; s > 0; s >>= 1) {
      if (t < s) {
        if (rv[t + s] > rv[t] || (rv[t + s] == rv[t] && ri[t + s] < ri[t])) {
          rv[t] = rv[t + s]; ri[t] = ri[t + s];
        }
      }
      __syncthreads();
    }
    if (t == 0) { s_last = ri[0]; o[it] = ri[0]; }
    __syncthreads();
  }
}

__global__ void gather_pts_kernel(const float* __restrict__ pts, const int* __restrict__ idx,
                                  float* __restrict__ out, int B, int N, int M) {
  int i = blockIdx.x * 256 + threadIdx.x;
  if (i >= B * M) return;
  int b = i / M;
  int id = idx[i];
  const float* s = pts + ((size_t)b * N + id) * 3;
  out[(size_t)i * 3 + 0] = s[0];
  out[(size_t)i * 3 + 1] = s[1];
  out[(size_t)i * 3 + 2] = s[2];
}

// first nsample hits in ascending index order, pad with first hit, all-zero if none.
__global__ void ball_query_kernel(const float* __restrict__ xyz, const float* __restrict__ nxyz,
                                  int* __restrict__ out, int B, int N, int M, int ns, float r2) {
  int i = blockIdx.x * 256 + threadIdx.x;
  if (i >= B * M) return;
  int b = i / M;
  const float* c = nxyz + (size_t)i * 3;
  const float* p = xyz + (size_t)b * N * 3;
  int* o = out + (size_t)i * ns;
  int cnt = 0, first = 0; bool any = false;
  for (int j = 0; j < N; ++j) {
    float dx = p[j * 3 + 0] - c[0], dy = p[j * 3 + 1] - c[1], dz = p[j * 3 + 2] - c[2];
    if (dx * dx + dy * dy + dz * dz < r2) {
      if (!any) { any = true; first = j; }
      if (cnt < ns) o[cnt++] = j;
      else break;
    }
  }
  int fill = any ? first : 0;
  for (int k = cnt; k < ns; ++k) o[k] = fill;
}

// x: (B, 3+Cf, M*ns): local xyz + gathered feats
__global__ void group_kernel(const float* __restrict__ xyz, const float* __restrict__ nxyz,
                             const float* __restrict__ feats, const int* __restrict__ gidx,
                             float* __restrict__ x, int B, int N, int M, int ns, int Cf) {
  int i = blockIdx.x * 256 + threadIdx.x;
  if (i >= B * M * ns) return;
  int s = i % ns; int m = (i / ns) % M; int b = i / (ns * M);
  int id = gidx[i];
  int Ct = 3 + Cf;
  size_t P = (size_t)M * ns;
  size_t p = (size_t)m * ns + s;
  const float* q = xyz + ((size_t)b * N + id) * 3;
  const float* c = nxyz + ((size_t)b * M + m) * 3;
  float* xb = x + (size_t)b * Ct * P;
  xb[0 * P + p] = q[0] - c[0];
  xb[1 * P + p] = q[1] - c[1];
  xb[2 * P + p] = q[2] - c[2];
  for (int cc = 0; cc < Cf; ++cc)
    xb[(size_t)(3 + cc) * P + p] = feats[((size_t)b * Cf + cc) * N + id];
}

// ---------------------------------------------------------------------------
// WMMA conv1x1 GEMM: y[b,o,p] = epi( sum_k W[o,k] * x[b,k,p] )
// mode 0: relu(v*g[o]+bb[o]);  mode 1: v + bb[o]
// Block 256 threads = 8 waves; block tile 32(O) x 64(P); K loop step 32.
// Fragments: A and B both use contiguous k = 16*(lane/16) + i (wave32).
// ---------------------------------------------------------------------------
__global__ __launch_bounds__(256) void conv_wmma_kernel(
    const float* __restrict__ W, const float* __restrict__ X,
    const float* __restrict__ G, const float* __restrict__ Bb,
    float* __restrict__ Y, int O, int K, int P, int mode) {
  __shared__ _Float16 As[32 * 32];
  __shared__ _Float16 Bs[32 * 64];
  int b = blockIdx.z;
  int o0 = blockIdx.y * 32;
  int p0 = blockIdx.x * 64;
  int t = threadIdx.x;
  int lane = t & 31;
  int w = t >> 5;
  int wo = w & 1;   // 2 o-tiles of 16
  int wp = w >> 1;  // 4 p-tiles of 16
  const float* Xb = X + (size_t)b * K * P;
  v8f acc = {};
  int m = lane & 15;
  int kbA = (lane < 16) ? 0 : 16;
  int n = lane & 15;
  int kbB = (lane < 16) ? 0 : 16;
  for (int k0 = 0; k0 < K; k0 += 32) {
    for (int e = t; e < 32 * 32; e += 256) {
      int oo = e >> 5, kk = e & 31;
      float v = 0.f;
      if (o0 + oo < O && k0 + kk < K) v = W[(size_t)(o0 + oo) * K + (k0 + kk)];
      As[oo * 32 + kk] = (_Float16)v;
    }
    for (int e = t; e < 32 * 64; e += 256) {
      int kk = e >> 6, pp = e & 63;
      float v = 0.f;
      if (k0 + kk < K && p0 + pp < P) v = Xb[(size_t)(k0 + kk) * P + (p0 + pp)];
      Bs[kk * 64 + pp] = (_Float16)v;
    }
    __syncthreads();
    v16h af, bf;
#pragma unroll
    for (int i = 0; i < 16; ++i) {
      int kk = kbA + i;
      af[i] = As[(wo * 16 + m) * 32 + kk];
    }
#pragma unroll
    for (int i = 0; i < 16; ++i) {
      bf[i] = Bs[(kbB + i) * 64 + wp * 16 + n];
    }
    acc = __builtin_amdgcn_wmma_f32_16x16x32_f16(false, af, false, bf,
                                                 (short)0, acc, false, false);
    __syncthreads();
  }
  int mb = (lane < 16) ? 0 : 8;
#pragma unroll
  for (int r = 0; r < 8; ++r) {
    int o = o0 + wo * 16 + mb + r;
    int p = p0 + wp * 16 + n;
    if (o < O && p < P) {
      float v = acc[r];
      if (mode == 0) { v = v * G[o] + Bb[o]; v = fmaxf(v, 0.f); }
      else           { v = v + Bb[o]; }
      Y[((size_t)b * O + o) * P + p] = v;
    }
  }
}

// y[b,c,j] = max_i x[b,c, j*jmul + i*imul]
__global__ void maxpool_kernel(const float* __restrict__ x, float* __restrict__ y,
                               int B, int C, int J, int cnt, int jmul, int imul, int Pin) {
  int i = blockIdx.x * 256 + threadIdx.x;
  if (i >= B * C * J) return;
  int j = i % J; int c = (i / J) % C; int b = i / (J * C);
  const float* xb = x + ((size_t)b * C + c) * Pin;
  float v = -1e30f;
  for (int k = 0; k < cnt; ++k) v = fmaxf(v, xb[(size_t)j * jmul + (size_t)k * imul]);
  y[((size_t)b * C + c) * J + j] = v;
}

// fusion (B, 4+F, n1*n2): ch0=cosine, ch1..3=txyz, ch4..=obj broadcast
__global__ void fusion_build_kernel(const float* __restrict__ tf, const float* __restrict__ sf,
                                    const float* __restrict__ txyz, float* __restrict__ fus,
                                    int B, int F, int n1, int n2) {
  int i = blockIdx.x * 256 + threadIdx.x;
  if (i >= B * n1 * n2) return;
  int c2 = i % n2; int a = (i / n2) % n1; int b = i / (n1 * n2);
  const float* tb = tf + (size_t)b * F * n1;
  const float* sb = sf + (size_t)b * F * n2;
  float num = 0.f, na = 0.f, nb = 0.f;
  for (int f = 0; f < F; ++f) {
    float u = tb[(size_t)f * n1 + a];
    float v = sb[(size_t)f * n2 + c2];
    num += u * v; na += u * u; nb += v * v;
  }
  float den = fmaxf(sqrtf(na) * sqrtf(nb), 1e-8f);
  int Ct = 4 + F;
  size_t P = (size_t)n1 * n2;
  size_t p = (size_t)a * n2 + c2;
  float* fb = fus + (size_t)b * Ct * P;
  fb[p] = num / den;
  fb[1 * P + p] = txyz[((size_t)b * n1 + a) * 3 + 0];
  fb[2 * P + p] = txyz[((size_t)b * n1 + a) * 3 + 1];
  fb[3 * P + p] = txyz[((size_t)b * n1 + a) * 3 + 2];
  for (int f = 0; f < F; ++f)
    fb[(size_t)(4 + f) * P + p] = tb[(size_t)f * n1 + a];
}

__global__ void top3_kernel(const float* __restrict__ unk, const float* __restrict__ kn,
                            int* __restrict__ idx3, float* __restrict__ w3,
                            int B, int U, int Kp) {
  int i = blockIdx.x * 256 + threadIdx.x;
  if (i >= B * U) return;
  int b = i / U;
  const float* u = unk + (size_t)i * 3;
  const float* k = kn + (size_t)b * Kp * 3;
  float d0 = 1e30f, d1 = 1e30f, d2 = 1e30f; int i0 = 0, i1 = 0, i2 = 0;
  for (int j = 0; j < Kp; ++j) {
    float dx = k[j * 3 + 0] - u[0], dy = k[j * 3 + 1] - u[1], dz = k[j * 3 + 2] - u[2];
    float d = dx * dx + dy * dy + dz * dz;
    if (d < d0)      { d2 = d1; i2 = i1; d1 = d0; i1 = i0; d0 = d; i0 = j; }
    else if (d < d1) { d2 = d1; i2 = i1; d1 = d; i1 = j; }
    else if (d < d2) { d2 = d; i2 = j; }
  }
  float w0 = 1.f / (d0 + 1e-8f), w1 = 1.f / (d1 + 1e-8f), w2 = 1.f / (d2 + 1e-8f);
  float s = w0 + w1 + w2;
  w3[(size_t)i * 3 + 0] = w0 / s; w3[(size_t)i * 3 + 1] = w1 / s; w3[(size_t)i * 3 + 2] = w2 / s;
  idx3[(size_t)i * 3 + 0] = i0; idx3[(size_t)i * 3 + 1] = i1; idx3[(size_t)i * 3 + 2] = i2;
}

__global__ void interp_kernel(const float* __restrict__ kf, const int* __restrict__ idx3,
                              const float* __restrict__ w3, float* __restrict__ out,
                              int B, int C, int U, int Np, int Cd, int doff) {
  int i = blockIdx.x * 256 + threadIdx.x;
  if (i >= B * C * U) return;
  int u = i % U; int c = (i / U) % C; int b = i / (U * C);
  const int* id = idx3 + ((size_t)b * U + u) * 3;
  const float* w = w3 + ((size_t)b * U + u) * 3;
  const float* f = kf + ((size_t)b * C + c) * Np;
  out[((size_t)b * Cd + doff + c) * U + u] =
      f[id[0]] * w[0] + f[id[1]] * w[1] + f[id[2]] * w[2];
}

__global__ void copy_ch_kernel(const float* __restrict__ src, int Cs, int soff, int C,
                               float* __restrict__ dst, int Cd, int doff, int P, int B) {
  int i = blockIdx.x * 256 + threadIdx.x;
  if (i >= B * C * P) return;
  int p = i % P; int c = (i / P) % C; int b = i / (P * C);
  dst[((size_t)b * Cd + doff + c) * P + p] = src[((size_t)b * Cs + soff + c) * P + p];
}

__global__ void xyzT_kernel(const float* __restrict__ src, float* __restrict__ dst,
                            int Cd, int P, int B) {
  int i = blockIdx.x * 256 + threadIdx.x;
  if (i >= B * P * 3) return;
  int k = i % 3; int p = (i / 3) % P; int b = i / (3 * P);
  dst[((size_t)b * Cd + k) * P + p] = src[((size_t)b * P + p) * 3 + k];
}

__global__ void extract_xyz_kernel(const float* __restrict__ v, float* __restrict__ out,
                                   int C, int P, int B) {
  int i = blockIdx.x * 256 + threadIdx.x;
  if (i >= B * P * 3) return;
  int k = i % 3; int p = (i / 3) % P; int b = i / (3 * P);
  out[((size_t)b * P + p) * 3 + k] = v[((size_t)b * C + k) * P + p];
}

__global__ void sigmoid_kernel(const float* __restrict__ x, float* __restrict__ y, int n) {
  int i = blockIdx.x * 256 + threadIdx.x;
  if (i < n) y[i] = 1.f / (1.f + expf(-x[i]));
}

__global__ void add_inplace_kernel(float* __restrict__ dst, const float* __restrict__ a, int n) {
  int i = blockIdx.x * 256 + threadIdx.x;
  if (i < n) dst[i] += a[i];
}

__global__ void copy_kernel(const float* __restrict__ s, float* __restrict__ d, int n) {
  int i = blockIdx.x * 256 + threadIdx.x;
  if (i < n) d[i] = s[i];
}

__global__ void boxes_kernel(const float* __restrict__ off, const float* __restrict__ centers,
                             float* __restrict__ out, int B, int M) {
  int i = blockIdx.x * 256 + threadIdx.x;
  if (i >= B * M) return;
  int b = i / M; int j = i % M;
  const float* ob = off + (size_t)b * 5 * M;
  float* o = out + (size_t)i * 5;
  o[0] = ob[0 * M + j] + centers[(size_t)i * 3 + 0];
  o[1] = ob[1 * M + j] + centers[(size_t)i * 3 + 1];
  o[2] = ob[2 * M + j] + centers[(size_t)i * 3 + 2];
  o[3] = ob[3 * M + j];
  o[4] = ob[4 * M + j];
}

__global__ void est_seg_kernel(const float* __restrict__ aux, const float* __restrict__ nf,
                               float* __restrict__ out, int B, int C, int N) {
  int i = blockIdx.x * 256 + threadIdx.x;
  if (i >= B * N) return;
  int b = i / N; int n = i % N;
  const float* f = nf + (size_t)b * C * N + n;
  float s = 0.f;
  for (int c = 0; c < C; ++c) s += aux[c] * f[(size_t)c * N];
  out[i] = s;
}

// ---------------------------------------------------------------------------
// Host-side orchestration
// ---------------------------------------------------------------------------
struct Mlp3 { const float* W[3]; const float* b[3]; const float* g[3]; int C[4]; };

struct Arena {
  char* base; size_t off;
  void* get(size_t bytes) {
    size_t a = (off + 255) & ~(size_t)255;
    off = a + bytes;
    return base + a;
  }
};

static void launch_conv(hipStream_t stream, const float* W, const float* X,
                        const float* G, const float* Bb, float* Y,
                        int O, int K, int P, int mode) {
  dim3 grid((P + 63) / 64, (O + 31) / 32, DEV_B);
  conv_wmma_kernel<<<grid, dim3(256), 0, stream>>>(W, X, G, Bb, Y, O, K, P, mode);
}

static void run_sa(hipStream_t stream, float* bufA, float* bufB,
                   const float* xyz, int N, const float* feats, int Cf,
                   int npoint, float radius, int ns, const Mlp3& m,
                   int* idxbuf, int* gidxbuf, float* new_xyz, float* out_feats) {
  const int B = DEV_B;
  fps_kernel<<<dim3(B), dim3(256), 0, stream>>>(xyz, N, npoint, idxbuf);
  int nGP = B * npoint;
  gather_pts_kernel<<<dim3((nGP + 255) / 256), dim3(256), 0, stream>>>(xyz, idxbuf, new_xyz, B, N, npoint);
  ball_query_kernel<<<dim3((nGP + 255) / 256), dim3(256), 0, stream>>>(xyz, new_xyz, gidxbuf, B, N, npoint, ns, radius * radius);
  int nG = B * npoint * ns;
  group_kernel<<<dim3((nG + 255) / 256), dim3(256), 0, stream>>>(xyz, new_xyz, feats, gidxbuf, bufA, B, N, npoint, ns, Cf);
  int P = npoint * ns;
  float* src = bufA; float* dst = bufB;
  for (int l = 0; l < 3; ++l) {
    launch_conv(stream, m.W[l], src, m.g[l], m.b[l], dst, m.C[l + 1], m.C[l], P, 0);
    float* t = src; src = dst; dst = t;
  }
  int nMP = B * m.C[3] * npoint;
  maxpool_kernel<<<dim3((nMP + 255) / 256), dim3(256), 0, stream>>>(src, out_feats, B, m.C[3], npoint, ns, ns, 1, P);
}

extern "C" void kernel_launch(void* const* d_in, const int* in_sizes, int n_in,
                              void* d_out, int out_size, void* d_ws, size_t ws_size,
                              hipStream_t stream) {
  const int B = DEV_B;
  const float* IN_T = (const float*)d_in[0];  // (32,512,3)
  const float* IN_S = (const float*)d_in[1];  // (32,1024,3)
#define PF(k) ((const float*)d_in[k])
  // params flattened jax-style (alphabetical keys; within conv: W,b,g or W,bias)
  const float* aux_seg = PF(2);
  const float *claW0 = PF(3), *clab0 = PF(4), *clag0 = PF(5);
  const float *claW1 = PF(6), *clab1 = PF(7), *clag1 = PF(8);
  const float *claW2 = PF(9), *claB2 = PF(10);
  const float *covW = PF(11), *covB = PF(12);
  const float *feaW0 = PF(13), *feab0 = PF(14), *feag0 = PF(15);
  const float *feaW1 = PF(16), *feaB1 = PF(17);
  const float *fp1W = PF(18), *fp1b = PF(19), *fp1g = PF(20);
  const float *fp2W = PF(21), *fp2b = PF(22), *fp2g = PF(23);
  const float *fp3W = PF(24), *fp3b = PF(25), *fp3g = PF(26);
  const float *prW0 = PF(27), *prb0 = PF(28), *prg0 = PF(29);
  const float *prW1 = PF(30), *prb1 = PF(31), *prg1 = PF(32);
  const float *prW2 = PF(33), *prB2 = PF(34);
  Mlp3 sa0 = {{PF(35), PF(38), PF(41)}, {PF(36), PF(39), PF(42)}, {PF(37), PF(40), PF(43)}, {3, 64, 64, 128}};
  Mlp3 sa1 = {{PF(44), PF(47), PF(50)}, {PF(45), PF(48), PF(51)}, {PF(46), PF(49), PF(52)}, {131, 128, 128, 256}};
  Mlp3 sa2 = {{PF(53), PF(56), PF(59)}, {PF(54), PF(57), PF(60)}, {PF(55), PF(58), PF(61)}, {259, 256, 256, 256}};
  const float *vtW0 = PF(62), *vtb0 = PF(63), *vtg0 = PF(64);
  const float *vtW1 = PF(65), *vtb1 = PF(66), *vtg1 = PF(67);
  const float *vtW2 = PF(68), *vtB2 = PF(69);
  Mlp3 vsa = {{PF(70), PF(73), PF(76)}, {PF(71), PF(74), PF(77)}, {PF(72), PF(75), PF(78)}, {260, 256, 256, 256}};
  Mlp3 xcm = {{PF(79), PF(82), PF(85)}, {PF(80), PF(83), PF(86)}, {PF(81), PF(84), PF(87)}, {260, 256, 256, 256}};
#undef PF

  float* out = (float*)d_out;
  float* out_cla = out;               // 32*128
  float* out_votexyz = out + 4096;    // 32*128*3
  float* out_boxes = out + 16384;     // 32*64*5
  float* out_centers = out + 26624;   // 32*64*3
  float* out_seg = out + 32768;       // 32*1024

  Arena ar{(char*)d_ws, 0};
  const size_t PP = (size_t)B * 260 * 8192;  // max activation elements
  float* bufA = (float*)ar.get(PP * 4);
  float* bufB = (float*)ar.get(PP * 4);
  // template persistents
  int* t_idx1 = (int*)ar.get((size_t)B * 256 * 4);
  float* t_xyz1 = (float*)ar.get((size_t)B * 256 * 3 * 4);
  int* t_gidx1 = (int*)ar.get((size_t)B * 256 * 32 * 4);
  float* t_feats1 = (float*)ar.get((size_t)B * 128 * 256 * 4);
  int* t_idx2 = (int*)ar.get((size_t)B * 128 * 4);
  float* t_xyz2 = (float*)ar.get((size_t)B * 128 * 3 * 4);
  int* t_gidx2 = (int*)ar.get((size_t)B * 128 * 32 * 4);
  float* t_feats2 = (float*)ar.get((size_t)B * 256 * 128 * 4);
  int* t_idx3 = (int*)ar.get((size_t)B * 64 * 4);
  float* t_xyz3 = (float*)ar.get((size_t)B * 64 * 3 * 4);
  int* t_gidx3 = (int*)ar.get((size_t)B * 64 * 32 * 4);
  float* t_feats3 = (float*)ar.get((size_t)B * 256 * 64 * 4);
  float* t_f = (float*)ar.get((size_t)B * 256 * 64 * 4);
  // search persistents
  int* s_idx1 = (int*)ar.get((size_t)B * 512 * 4);
  float* s_xyz1 = (float*)ar.get((size_t)B * 512 * 3 * 4);
  int* s_gidx1 = (int*)ar.get((size_t)B * 512 * 32 * 4);
  float* s_feats1 = (float*)ar.get((size_t)B * 128 * 512 * 4);
  int* s_idx2 = (int*)ar.get((size_t)B * 256 * 4);
  float* s_xyz2 = (float*)ar.get((size_t)B * 256 * 3 * 4);
  int* s_gidx2 = (int*)ar.get((size_t)B * 256 * 32 * 4);
  float* s_feats2 = (float*)ar.get((size_t)B * 256 * 256 * 4);
  int* s_idx3 = (int*)ar.get((size_t)B * 128 * 4);
  float* s_xyz3 = (float*)ar.get((size_t)B * 128 * 3 * 4);
  int* s_gidx3 = (int*)ar.get((size_t)B * 128 * 32 * 4);
  float* s_feats3 = (float*)ar.get((size_t)B * 256 * 128 * 4);
  float* s_f = (float*)ar.get((size_t)B * 256 * 128 * 4);
  // xcorr / heads
  float* fus_pool = (float*)ar.get((size_t)B * 256 * 128 * 4);
  float* fea_t = (float*)ar.get((size_t)B * 256 * 128 * 4);
  float* fusion_f = (float*)ar.get((size_t)B * 256 * 128 * 4);
  float* ca = (float*)ar.get((size_t)B * 256 * 128 * 4);
  float* cb = (float*)ar.get((size_t)B * 256 * 128 * 4);
  float* score = (float*)ar.get((size_t)B * 128 * 4);
  float* fxf = (float*)ar.get((size_t)B * 259 * 128 * 4);
  float* va = (float*)ar.get((size_t)B * 256 * 128 * 4);
  float* vb = (float*)ar.get((size_t)B * 256 * 128 * 4);
  float* vc = (float*)ar.get((size_t)B * 259 * 128 * 4);
  float* vx = (float*)ar.get((size_t)B * 128 * 3 * 4);
  float* vf = (float*)ar.get((size_t)B * 257 * 128 * 4);
  int* vidx = (int*)ar.get((size_t)B * 64 * 4);
  int* vgidx = (int*)ar.get((size_t)B * 64 * 16 * 4);
  float* centers = (float*)ar.get((size_t)B * 64 * 3 * 4);
  float* prop_feat = (float*)ar.get((size_t)B * 256 * 64 * 4);
  float* pa = (float*)ar.get((size_t)B * 256 * 64 * 4);
  float* pb = (float*)ar.get((size_t)B * 256 * 64 * 4);
  float* offb = (float*)ar.get((size_t)B * 5 * 64 * 4);
  int* idx3a = (int*)ar.get((size_t)B * 256 * 3 * 4);
  float* w3a = (float*)ar.get((size_t)B * 256 * 3 * 4);
  int* idx3b = (int*)ar.get((size_t)B * 512 * 3 * 4);
  float* w3b = (float*)ar.get((size_t)B * 512 * 3 * 4);
  int* idx3c = (int*)ar.get((size_t)B * 1024 * 3 * 4);
  float* w3c = (float*)ar.get((size_t)B * 1024 * 3 * 4);
  float* nf1 = (float*)ar.get((size_t)B * 256 * 256 * 4);
  float* nf2 = (float*)ar.get((size_t)B * 256 * 512 * 4);

#define G1(n) dim3((((n) + 255) / 256)), dim3(256), 0, stream

  // ---------------- template backbone ----------------
  run_sa(stream, bufA, bufB, IN_T, 512, nullptr, 0, 256, 0.3f, 32, sa0, t_idx1, t_gidx1, t_xyz1, t_feats1);
  run_sa(stream, bufA, bufB, t_xyz1, 256, t_feats1, 128, 128, 0.5f, 32, sa1, t_idx2, t_gidx2, t_xyz2, t_feats2);
  run_sa(stream, bufA, bufB, t_xyz2, 128, t_feats2, 256, 64, 0.7f, 32, sa2, t_idx3, t_gidx3, t_xyz3, t_feats3);
  launch_conv(stream, covW, t_feats3, nullptr, covB, t_f, 256, 256, 64, 1);

  // ---------------- search backbone ----------------
  run_sa(stream, bufA, bufB, IN_S, 1024, nullptr, 0, 512, 0.3f, 32, sa0, s_idx1, s_gidx1, s_xyz1, s_feats1);
  run_sa(stream, bufA, bufB, s_xyz1, 512, s_feats1, 128, 256, 0.5f, 32, sa1, s_idx2, s_gidx2, s_xyz2, s_feats2);
  run_sa(stream, bufA, bufB, s_xyz2, 256, s_feats2, 256, 128, 0.7f, 32, sa2, s_idx3, s_gidx3, s_xyz3, s_feats3);
  launch_conv(stream, covW, s_feats3, nullptr, covB, s_f, 256, 256, 128, 1);

  // ---------------- xcorr fusion ----------------
  fusion_build_kernel<<<G1(B * 64 * 128)>>>(t_f, s_f, t_xyz3, bufA, B, 256, 64, 128);
  launch_conv(stream, xcm.W[0], bufA, xcm.g[0], xcm.b[0], bufB, 256, 260, 8192, 0);
  launch_conv(stream, xcm.W[1], bufB, xcm.g[1], xcm.b[1], bufA, 256, 256, 8192, 0);
  launch_conv(stream, xcm.W[2], bufA, xcm.g[2], xcm.b[2], bufB, 256, 256, 8192, 0);
  maxpool_kernel<<<G1(B * 256 * 128)>>>(bufB, fus_pool, B, 256, 128, 64, 1, 128, 8192);
  launch_conv(stream, feaW0, fus_pool, feag0, feab0, fea_t, 256, 256, 128, 0);
  launch_conv(stream, feaW1, fea_t, nullptr, feaB1, fusion_f, 256, 256, 128, 1);

  // ---------------- classification head ----------------
  launch_conv(stream, claW0, fusion_f, clag0, clab0, ca, 256, 256, 128, 0);
  launch_conv(stream, claW1, ca, clag1, clab1, cb, 256, 256, 128, 0);
  launch_conv(stream, claW2, cb, nullptr, claB2, out_cla, 1, 256, 128, 1);
  sigmoid_kernel<<<G1(B * 128)>>>(out_cla, score, B * 128);

  // ---------------- vote ----------------
  xyzT_kernel<<<G1(B * 128 * 3)>>>(s_xyz3, fxf, 259, 128, B);
  copy_ch_kernel<<<G1(B * 256 * 128)>>>(fusion_f, 256, 0, 256, fxf, 259, 3, 128, B);
  launch_conv(stream, vtW0, fxf, vtg0, vtb0, va, 256, 259, 128, 0);
  launch_conv(stream, vtW1, va, vtg1, vtb1, vb, 256, 256, 128, 0);
  launch_conv(stream, vtW2, vb, nullptr, vtB2, vc, 259, 256, 128, 1);
  add_inplace_kernel<<<G1(B * 259 * 128)>>>(vc, fxf, B * 259 * 128);
  extract_xyz_kernel<<<G1(B * 128 * 3)>>>(vc, vx, 259, 128, B);
  copy_kernel<<<G1(B * 128 * 3)>>>(vx, out_votexyz, B * 128 * 3);
  copy_ch_kernel<<<G1(B * 1 * 128)>>>(score, 1, 0, 1, vf, 257, 0, 128, B);
  copy_ch_kernel<<<G1(B * 256 * 128)>>>(vc, 259, 3, 256, vf, 257, 1, 128, B);

  // ---------------- vote SA + proposal ----------------
  run_sa(stream, bufA, bufB, vx, 128, vf, 257, 64, 0.3f, 16, vsa, vidx, vgidx, centers, prop_feat);
  copy_kernel<<<G1(B * 64 * 3)>>>(centers, out_centers, B * 64 * 3);
  launch_conv(stream, prW0, prop_feat, prg0, prb0, pa, 256, 256, 64, 0);
  launch_conv(stream, prW1, pa, prg1, prb1, pb, 256, 256, 64, 0);
  launch_conv(stream, prW2, pb, nullptr, prB2, offb, 5, 256, 64, 1);
  boxes_kernel<<<G1(B * 64)>>>(offb, centers, out_boxes, B, 64);

  // ---------------- feature propagation ----------------
  top3_kernel<<<G1(B * 256)>>>(s_xyz2, s_xyz3, idx3a, w3a, B, 256, 128);
  interp_kernel<<<G1(B * 256 * 256)>>>(s_f, idx3a, w3a, bufA, B, 256, 256, 128, 512, 0);
  copy_ch_kernel<<<G1(B * 256 * 256)>>>(s_feats2, 256, 0, 256, bufA, 512, 256, 256, B);
  launch_conv(stream, fp1W, bufA, fp1g, fp1b, nf1, 256, 512, 256, 0);

  top3_kernel<<<G1(B * 512)>>>(s_xyz1, s_xyz2, idx3b, w3b, B, 512, 256);
  interp_kernel<<<G1(B * 256 * 512)>>>(nf1, idx3b, w3b, bufA, B, 256, 512, 256, 384, 0);
  copy_ch_kernel<<<G1(B * 128 * 512)>>>(s_feats1, 128, 0, 128, bufA, 384, 256, 512, B);
  launch_conv(stream, fp2W, bufA, fp2g, fp2b, nf2, 256, 384, 512, 0);

  top3_kernel<<<G1(B * 1024)>>>(IN_S, s_xyz1, idx3c, w3c, B, 1024, 512);
  interp_kernel<<<G1(B * 256 * 1024)>>>(nf2, idx3c, w3c, bufA, B, 256, 1024, 512, 256, 0);
  launch_conv(stream, fp3W, bufA, fp3g, fp3b, bufB, 256, 256, 1024, 0);
  est_seg_kernel<<<G1(B * 1024)>>>(aux_seg, bufB, out_seg, B, 256, 1024);

#undef G1
  (void)in_sizes; (void)n_in; (void)out_size; (void)ws_size;
}
